// MultiHeadAttention_67121748902329
// MI455X (gfx1250) — matrix operand order repaired and checked by reading, not compiled, not run on hardware
//
#include <hip/hip_runtime.h>
#include <hip/hip_bf16.h>

#ifndef __has_builtin
#define __has_builtin(x) 0
#endif

typedef __attribute__((ext_vector_type(16))) _Float16 v16h;
typedef __attribute__((ext_vector_type(8)))  _Float16 v8h;
typedef __attribute__((ext_vector_type(8)))  float    v8f;
typedef __attribute__((ext_vector_type(4)))  int      v4i;

#define MHA_B 2
#define MHA_S 2048
#define MHA_D 1024
#define MHA_H 16
#define MHA_HD 64

#if __has_builtin(__builtin_amdgcn_global_load_async_to_lds_b128)
#define USE_ASYNC_LDS 1
#endif
#if __has_builtin(__builtin_amdgcn_update_dpp)
#define USE_DPP_RED 1
#endif

__device__ __forceinline__ void wait_async0() {
#if __has_builtin(__builtin_amdgcn_s_wait_asynccnt)
  __builtin_amdgcn_s_wait_asynccnt(0);
#else
  asm volatile("s_wait_asynccnt 0x0" ::: "memory");
#endif
}
__device__ __forceinline__ void wait_async1() {
#if __has_builtin(__builtin_amdgcn_s_wait_asynccnt)
  __builtin_amdgcn_s_wait_asynccnt(1);
#else
  asm volatile("s_wait_asynccnt 0x1" ::: "memory");
#endif
}

// ---- 16-lane all-reduce via DPP16 (no LDS, no dscnt stalls) ----------------
#ifdef USE_DPP_RED
template <int CTRL>
__device__ __forceinline__ float dpp_mov_f32(float x) {
  int xi = __builtin_bit_cast(int, x);
  int r  = __builtin_amdgcn_update_dpp(xi, xi, CTRL, 0xF, 0xF, true);
  return __builtin_bit_cast(float, r);
}
__device__ __forceinline__ float red16_max(float v) {
  v = fmaxf(v, dpp_mov_f32<0xB1>(v));   // quad_perm(1,0,3,2)  : xor1
  v = fmaxf(v, dpp_mov_f32<0x4E>(v));   // quad_perm(2,3,0,1)  : xor2
  v = fmaxf(v, dpp_mov_f32<0x141>(v));  // row_half_mirror     : 8-group
  v = fmaxf(v, dpp_mov_f32<0x140>(v));  // row_mirror          : 16-lane
  return v;
}
__device__ __forceinline__ float red16_sum(float v) {
  v += dpp_mov_f32<0xB1>(v);
  v += dpp_mov_f32<0x4E>(v);
  v += dpp_mov_f32<0x141>(v);
  v += dpp_mov_f32<0x140>(v);
  return v;
}
#else
__device__ __forceinline__ float red16_max(float v) {
  for (int off = 1; off < 16; off <<= 1) v = fmaxf(v, __shfl_xor(v, off, 32));
  return v;
}
__device__ __forceinline__ float red16_sum(float v) {
  for (int off = 1; off < 16; off <<= 1) v += __shfl_xor(v, off, 32);
  return v;
}
#endif

__device__ __forceinline__ v8f wmma_f16f32(v16h a, v16h b, v8f c) {
  // D = A(16x32 f16) * B(32x16 f16) + C(16x16 f32)
  return __builtin_amdgcn_wmma_f32_16x16x32_f16(
      /*neg_a=*/false, a, /*neg_b=*/false, b,
      /*c_mod=*/(short)0, c, /*reuse_a=*/false, /*reuse_b=*/false);
}

// ---- fragment loaders (per ISA 16-bit layouts, wave32) --------------------

// A fragment: 16x32, element (m,k). lane: m = lane&15, k-base = +8 for upper half.
// VGPR v holds K pair: k = kb + 2*(v%4) + (v>=4 ? 16 : 0)
// f16 source: two contiguous 16B runs per lane -> 2x (global|ds)_load_b128.
template <typename T>
__device__ __forceinline__ v16h load_A_tile(const T* base, int ld, int m0, int k0, int lane) {
  const T* row = base + (size_t)(m0 + (lane & 15)) * ld;
  int kb = k0 + ((lane >> 4) << 3);
  v16h a;
#pragma unroll
  for (int v = 0; v < 8; ++v) {
    int kk = kb + ((v & 3) << 1) + ((v & 4) << 2);
    a[2 * v]     = (_Float16)row[kk];
    a[2 * v + 1] = (_Float16)row[kk + 1];
  }
  return a;
}

// B fragment: 32x16, element (kk,n) from an n-major matrix:
// B(kk,n) = base[(row0+n)*ld + k0+kk]. lane: n = lane&15, kk = 8*(lane>>4)+h+(h&8).
// Per-lane read is two 8-f16 (16B) runs -> 2x global_load_b128.
__device__ __forceinline__ v16h load_Bt_tile(const _Float16* base, int ld, int row0, int k0, int lane) {
  const _Float16* p = base + (size_t)(row0 + (lane & 15)) * ld + k0 + ((lane >> 4) << 3);
  v16h b;
#pragma unroll
  for (int h = 0; h < 16; ++h) b[h] = p[h + (h & 8)];
  return b;
}

// ---- prep: X f32 -> f16 ----------------------------------------------------
__global__ void conv_f32_to_f16(const float* __restrict__ X, _Float16* __restrict__ Xh, int n) {
  int i = (blockIdx.x * blockDim.x + threadIdx.x) * 8;
  if (i >= n) return;
  float4 a = *reinterpret_cast<const float4*>(X + i);
  float4 b = *reinterpret_cast<const float4*>(X + i + 4);
  v8h h;
  h[0] = (_Float16)a.x; h[1] = (_Float16)a.y; h[2] = (_Float16)a.z; h[3] = (_Float16)a.w;
  h[4] = (_Float16)b.x; h[5] = (_Float16)b.y; h[6] = (_Float16)b.z; h[7] = (_Float16)b.w;
  *reinterpret_cast<v8h*>(Xh + i) = h;
}

// ---- prep: Wt[n*K + k] = (f16) W[k*N + n] ----------------------------------
__global__ void transpose_conv_f16(const float* __restrict__ W, _Float16* __restrict__ Wt,
                                   int K, int N) {
  __shared__ float tile[32][33];
  int tx = threadIdx.x & 31, ty = threadIdx.x >> 5;  // 32 x 8
  int k0 = blockIdx.x * 32, n0 = blockIdx.y * 32;
#pragma unroll
  for (int i = ty; i < 32; i += 8)
    tile[i][tx] = W[(size_t)(k0 + i) * N + n0 + tx];
  __syncthreads();
#pragma unroll
  for (int i = ty; i < 32; i += 8)
    Wt[(size_t)(n0 + i) * K + k0 + tx] = (_Float16)tile[tx][i];
}

// ---- async (or sync) staging of a 16x32-half A tile into LDS ---------------
// 16 rows x 64B = 1KB; threads 0..63 each move one 16B chunk.
__device__ __forceinline__ void stage_A(const _Float16* A, int K, int m0, int k0,
                                        _Float16* dstbuf, int tid) {
  if (tid < 64) {
    const int row = tid >> 2, chunk = tid & 3;
    _Float16* src = const_cast<_Float16*>(A) + (size_t)(m0 + row) * K + k0 + chunk * 8;
    _Float16* dst = dstbuf + row * 32 + chunk * 8;
#ifdef USE_ASYNC_LDS
    __builtin_amdgcn_global_load_async_to_lds_b128(
        (__attribute__((address_space(1))) v4i*)src,
        (__attribute__((address_space(3))) v4i*)dst, 0, 0);
#else
    *reinterpret_cast<uint4*>(dst) = *reinterpret_cast<const uint4*>(src);
#endif
  }
}

// ---- GEMM: C[MxN] = A[MxK](f16) @ W[KxN] + bias; W passed transposed f16 ---
// 4 waves/block -> 16x256 of C; A tile double-buffered in LDS via async loads.
// TRANS_OUT: store C transposed as VT[(m>>11)*1024 + n][m&2047] (f16, packed 16B).
template <typename TOUT, bool TRANS_OUT>
__global__ void wmma_gemm_bias(const _Float16* __restrict__ A, const _Float16* __restrict__ Wt,
                               const float* __restrict__ bias, TOUT* __restrict__ C,
                               int M, int N, int K) {
  const int tid  = threadIdx.x;
  const int lane = tid & 31;
  const int wave = tid >> 5;
  const int m0 = blockIdx.x << 4;
  const int n0 = (blockIdx.y << 8) + (wave << 6);

  __shared__ __align__(16) _Float16 Ash[2][16 * 32];

  stage_A(A, K, m0, 0, Ash[0], tid);

  v8f acc0 = {}, acc1 = {}, acc2 = {}, acc3 = {};
  for (int k0 = 0, it = 0; k0 < K; k0 += 32, ++it) {
    const int buf = it & 1;
    const bool more = (k0 + 32 < K);
    if (more) {
      stage_A(A, K, m0, k0 + 32, Ash[buf ^ 1], tid);  // prefetch next tile (async)
      wait_async1();                                  // current tile resident
    } else {
      wait_async0();
    }
    __syncthreads();

    v16h a  = load_A_tile(Ash[buf], 32, 0, 0, lane);  // 2x ds_load_b128
    v16h b0 = load_Bt_tile(Wt, K, n0,      k0, lane);
    v16h b1 = load_Bt_tile(Wt, K, n0 + 16, k0, lane);
    v16h b2 = load_Bt_tile(Wt, K, n0 + 32, k0, lane);
    v16h b3 = load_Bt_tile(Wt, K, n0 + 48, k0, lane);
    if (more) {  // global_prefetch_b8 for next W k-step
      __builtin_prefetch(Wt + (size_t)(n0 + (lane & 15)) * K + k0 + 32, 0, 3);
      __builtin_prefetch(Wt + (size_t)(n0 + 32 + (lane & 15)) * K + k0 + 32, 0, 3);
    }
    acc0 = wmma_f16f32(a, b0, acc0);
    acc1 = wmma_f16f32(a, b1, acc1);
    acc2 = wmma_f16f32(a, b2, acc2);
    acc3 = wmma_f16f32(a, b3, acc3);
    __syncthreads();
  }

  const int n = lane & 15;
  const int mrow = m0 + ((lane >> 4) << 3);
  const float bb[4] = {bias[n0 + n], bias[n0 + 16 + n], bias[n0 + 32 + n], bias[n0 + 48 + n]};
  v8f accs[4] = {acc0, acc1, acc2, acc3};

  if (TRANS_OUT) {
    // VT[( (m>>11)*1024 + ncol ) * 2048 + (m & 2047)]; 8 consecutive m -> 16B run
#pragma unroll
    for (int t = 0; t < 4; ++t) {
      const int ncol = n0 + t * 16 + n;
      v8h pack;
#pragma unroll
      for (int r = 0; r < 8; ++r) pack[r] = (_Float16)(accs[t][r] + bb[t]);
      const size_t base = ((size_t)(mrow >> 11) * 1024 + ncol) * 2048 + (mrow & 2047);
      *reinterpret_cast<v8h*>((_Float16*)C + base) = pack;
    }
  } else {
#pragma unroll
    for (int r = 0; r < 8; ++r) {
      const size_t row = (size_t)(mrow + r) * N;
#pragma unroll
      for (int t = 0; t < 4; ++t)
        C[row + n0 + t * 16 + n] = (TOUT)(accs[t][r] + bb[t]);
    }
  }
}

// ---- causal flash attention: one wave per (b, h, 16-query tile) ------------
// Q,K row-major [B,S,D]; V pre-transposed: VT[b*1024 + h*64 + d][s] (f16).
__global__ void attn_flash(const _Float16* __restrict__ Q, const _Float16* __restrict__ K,
                           const _Float16* __restrict__ VT, _Float16* __restrict__ O) {
  const int lane = threadIdx.x;
  const int qt = blockIdx.x, h = blockIdx.y, b = blockIdx.z;
  const int D = MHA_D, S = MHA_S;
  const int qb = qt << 4;

  const _Float16* Qp  = Q  + (size_t)b * S * D + h * MHA_HD;
  const _Float16* Kp  = K  + (size_t)b * S * D + h * MHA_HD;
  const _Float16* Vtp = VT + ((size_t)b * 1024 + h * MHA_HD) * S;

  __shared__ __align__(16) _Float16 Pt[16 * 32];  // P tile (C-layout -> A-layout transit)

  // Q tile as two A fragments (k = 0..31 and 32..63)
  v16h aq0 = load_A_tile(Qp, D, qb, 0, lane);
  v16h aq1 = load_A_tile(Qp, D, qb, 32, lane);

  v8f o0 = {}, o1 = {}, o2 = {}, o3 = {};
  float mrow[8], lrow[8];
#pragma unroll
  for (int r = 0; r < 8; ++r) { mrow[r] = -INFINITY; lrow[r] = 0.f; }

  const float scale = 0.125f;  // 1/sqrt(64)
  const int qrow_off = ((lane >> 4) << 3);
  const int nkb = (qb + 16 + 31) >> 5;  // 32-key blocks under causal mask

  for (int kb = 0; kb < nkb; ++kb) {
    const int key0 = kb << 5;

    // ---- scores S = Q * K^T for 32 keys (two 16-col C tiles) ----
    v8f s0 = {}, s1 = {};
    {
      v16h b0 = load_Bt_tile(Kp, D, key0,      0,  lane);
      v16h b1 = load_Bt_tile(Kp, D, key0,      32, lane);
      s0 = wmma_f16f32(aq0, b0, s0);
      s0 = wmma_f16f32(aq1, b1, s0);
      v16h b2 = load_Bt_tile(Kp, D, key0 + 16, 0,  lane);
      v16h b3 = load_Bt_tile(Kp, D, key0 + 16, 32, lane);
      s1 = wmma_f16f32(aq0, b2, s1);
      s1 = wmma_f16f32(aq1, b3, s1);
    }
    if (kb + 1 < nkb) {  // prefetch next key block (K rows + VT rows)
      __builtin_prefetch(Kp + (size_t)(key0 + 32 + lane) * D, 0, 3);
      __builtin_prefetch(Vtp + (size_t)lane * S + key0 + 32, 0, 3);
      __builtin_prefetch(Vtp + (size_t)(lane + 32) * S + key0 + 32, 0, 3);
    }

    // ---- causal mask + online softmax (row = qb + qrow_off + r) ----
    const int kcol0 = key0 + (lane & 15);
    const int kcol1 = kcol0 + 16;
#pragma unroll
    for (int r = 0; r < 8; ++r) {
      const int q = qb + qrow_off + r;
      float v0 = (kcol0 <= q) ? s0[r] * scale : -INFINITY;
      float v1 = (kcol1 <= q) ? s1[r] * scale : -INFINITY;
      const float rmax = red16_max(fmaxf(v0, v1));
      const float mnew = fmaxf(mrow[r], rmax);
      const float p0 = (kcol0 <= q) ? __expf(v0 - mnew) : 0.f;
      const float p1 = (kcol1 <= q) ? __expf(v1 - mnew) : 0.f;
      const float rsum = red16_sum(p0 + p1);
      const float corr = __expf(mrow[r] - mnew);  // 0 when mrow == -inf
      lrow[r] = lrow[r] * corr + rsum;
      mrow[r] = mnew;
      o0[r] *= corr; o1[r] *= corr; o2[r] *= corr; o3[r] *= corr;
      s0[r] = p0; s1[r] = p1;
    }

    // ---- P: C-layout -> LDS -> A-layout fragment ----
#pragma unroll
    for (int r = 0; r < 8; ++r) {
      const int m = qrow_off + r;
      Pt[m * 32 + (lane & 15)]      = (_Float16)s0[r];
      Pt[m * 32 + 16 + (lane & 15)] = (_Float16)s1[r];
    }
    __syncthreads();
    v16h ap = load_A_tile(Pt, 32, 0, 0, lane);  // 2x ds_load_b128

    // ---- O += P * V : B fragments straight from VT (2x global_load_b128) ----
    v16h bv0 = load_Bt_tile(Vtp, S, 0,  key0, lane);
    v16h bv1 = load_Bt_tile(Vtp, S, 16, key0, lane);
    v16h bv2 = load_Bt_tile(Vtp, S, 32, key0, lane);
    v16h bv3 = load_Bt_tile(Vtp, S, 48, key0, lane);
    o0 = wmma_f16f32(ap, bv0, o0);
    o1 = wmma_f16f32(ap, bv1, o1);
    o2 = wmma_f16f32(ap, bv2, o2);
    o3 = wmma_f16f32(ap, bv3, o3);
    __syncthreads();
  }

  // ---- normalize + store (layout [B,S,D], D index = h*64 + d) ----
  _Float16* Op = O + (size_t)b * S * D + h * MHA_HD;
  const int n = lane & 15;
#pragma unroll
  for (int r = 0; r < 8; ++r) {
    const int q = qb + qrow_off + r;
    const float inv = 1.0f / lrow[r];
    Op[(size_t)q * D + n]      = (_Float16)(o0[r] * inv);
    Op[(size_t)q * D + 16 + n] = (_Float16)(o1[r] * inv);
    Op[(size_t)q * D + 32 + n] = (_Float16)(o2[r] * inv);
    Op[(size_t)q * D + 48 + n] = (_Float16)(o3[r] * inv);
  }
}

extern "C" void kernel_launch(void* const* d_in, const int* in_sizes, int n_in,
                              void* d_out, int out_size, void* d_ws, size_t ws_size,
                              hipStream_t stream) {
  (void)in_sizes; (void)n_in; (void)out_size; (void)ws_size;
  const float* X  = (const float*)d_in[0];
  const float* Wq = (const float*)d_in[1];
  const float* bq = (const float*)d_in[2];
  const float* Wk = (const float*)d_in[3];
  const float* bk = (const float*)d_in[4];
  const float* Wv = (const float*)d_in[5];
  const float* bv = (const float*)d_in[6];
  const float* Wo = (const float*)d_in[7];
  const float* bo = (const float*)d_in[8];
  float* out = (float*)d_out;

  const int B = MHA_B, S = MHA_S, D = MHA_D;
  const int M = B * S;
  const size_t WSZ = (size_t)D * D;   // weight elems
  const size_t ASZ = (size_t)M * D;   // activation elems

  _Float16* Wqt = (_Float16*)d_ws;
  _Float16* Wkt = Wqt + WSZ;
  _Float16* Wvt = Wkt + WSZ;
  _Float16* Wot = Wvt + WSZ;
  _Float16* Xh  = Wot + WSZ;
  _Float16* Qh  = Xh + ASZ;
  _Float16* Kh  = Qh + ASZ;
  _Float16* VTh = Kh + ASZ;   // V stored transposed: [b*1024 + n][s]
  _Float16* Ah  = VTh + ASZ;

  // 1) prep: transpose+convert weights; convert X to f16
  dim3 tgrid(D / 32, D / 32);
  transpose_conv_f16<<<tgrid, 256, 0, stream>>>(Wq, Wqt, D, D);
  transpose_conv_f16<<<tgrid, 256, 0, stream>>>(Wk, Wkt, D, D);
  transpose_conv_f16<<<tgrid, 256, 0, stream>>>(Wv, Wvt, D, D);
  transpose_conv_f16<<<tgrid, 256, 0, stream>>>(Wo, Wot, D, D);
  conv_f32_to_f16<<<(M * D) / (256 * 8), 256, 0, stream>>>(X, Xh, M * D);

  // 2) QKV projections (f16 X -> f16 Q/K row-major, V transposed)
  dim3 ggrid(M / 16, D / 256);
  wmma_gemm_bias<_Float16, false><<<ggrid, 128, 0, stream>>>(Xh, Wqt, bq, Qh, M, D, D);
  wmma_gemm_bias<_Float16, false><<<ggrid, 128, 0, stream>>>(Xh, Wkt, bk, Kh, M, D, D);
  wmma_gemm_bias<_Float16, true ><<<ggrid, 128, 0, stream>>>(Xh, Wvt, bv, VTh, M, D, D);

  // 3) causal flash attention
  dim3 agrid(S / 16, MHA_H, B);
  attn_flash<<<agrid, 32, 0, stream>>>(Qh, Kh, VTh, Ah);

  // 4) output projection (f16 attn -> f32 out)
  wmma_gemm_bias<float, false><<<ggrid, 128, 0, stream>>>(Ah, Wot, bo, out, M, D, D);
}
